// DA_Block_62474594287831
// MI455X (gfx1250) — compile-verified, matrix operand order checked by reading, not run on hardware
//
#include <hip/hip_runtime.h>
#include <hip/hip_bf16.h>

// ---------------------------------------------------------------------------
// DA block (dual attention) for MI455X / gfx1250, wave32 + WMMA bf16.
// B=2, C=256, H=W=D=16 -> N=4096, CQK=32.
// All GEMMs via v_wmma_f32_16x16x32_bf16; PAM attention fused (flash-style).
// ---------------------------------------------------------------------------

typedef __bf16 bf16_t;
typedef __attribute__((ext_vector_type(16))) __bf16 v16bf;
typedef __attribute__((ext_vector_type(8)))  __bf16 v8bf;
typedef __attribute__((ext_vector_type(8)))  float  v8f;

constexpr int kB   = 2;
constexpr int kC   = 256;
constexpr int kN   = 4096;
constexpr int kCQK = 32;

// ---------------------------------------------------------------------------
// Fragment loaders.
//
// bf16 A-matrix 16x32 layout (ISA 7.12.2): lane m in [0,16) holds M=m with
// K = {0..7, 16..23}; lane m+16 holds M=m with K = {8..15, 24..31}; K packed
// in pairs per VGPR => v16bf element i maps to K = (halfsel*8) + (i<8 ? i : 8+i).
// The B-matrix 32x16 layout is the mirrored striping with lane selecting the
// N column.  Both loaders below assume the selected line (row of A, or the
// K-major line backing a B column) is contiguous in K in memory.
// ---------------------------------------------------------------------------
__device__ inline v16bf load_frag_k32(const bf16_t* lineptr, int lane) {
  int ko = (lane & 16) ? 8 : 0;
  v8bf lo = *(const v8bf*)(lineptr + ko);        // K = ko .. ko+7
  v8bf hi = *(const v8bf*)(lineptr + ko + 16);   // K = ko+16 .. ko+23
  v16bf f;
  #pragma unroll
  for (int i = 0; i < 8; ++i) { f[i] = lo[i]; f[i + 8] = hi[i]; }
  return f;
}

__device__ inline v16bf load_frag_k32_f32(const float* lineptr, int lane) {
  int ko = (lane & 16) ? 8 : 0;
  v16bf f;
  #pragma unroll
  for (int i = 0; i < 8; ++i) {
    f[i]     = (bf16_t)lineptr[ko + i];
    f[i + 8] = (bf16_t)lineptr[ko + 16 + i];
  }
  return f;
}

__device__ inline v8f wmma_bf16(v16bf a, v16bf b, v8f c) {
  // (neg_a, A, neg_b, B, c_mod, C, reuse_a, reuse_b)
  return __builtin_amdgcn_wmma_f32_16x16x32_bf16(false, a, false, b,
                                                 (short)0, c, false, false);
}

// Reductions across the 16-lane half groups (C/D tile columns live in lanes).
__device__ inline float rmax16(float v) {
  #pragma unroll
  for (int m = 1; m < 16; m <<= 1) v = fmaxf(v, __shfl_xor(v, m, 32));
  return v;
}
__device__ inline float rsum16(float v) {
  #pragma unroll
  for (int m = 1; m < 16; m <<= 1) v += __shfl_xor(v, m, 32);
  return v;
}
__device__ inline float rmin32(float v) {
  #pragma unroll
  for (int m = 1; m < 32; m <<= 1) v = fminf(v, __shfl_xor(v, m, 32));
  return v;
}
__device__ inline float rsum32(float v) {
  #pragma unroll
  for (int m = 1; m < 32; m <<= 1) v += __shfl_xor(v, m, 32);
  return v;
}

// ---------------------------------------------------------------------------
// K1: f32 -> bf16 copies: xb = (B,C,N), xbT = (B,N,C)
// ---------------------------------------------------------------------------
__global__ void convert_kernel(const float* __restrict__ x,
                               bf16_t* __restrict__ xb,
                               bf16_t* __restrict__ xbT) {
  int idx = blockIdx.x * blockDim.x + threadIdx.x;
  if (idx >= kB * kC * kN) return;
  int n = idx & (kN - 1);
  int c = (idx >> 12) & (kC - 1);
  int b = idx >> 20;
  float v = x[idx];
  xb[idx] = (bf16_t)v;
  xbT[((size_t)(b * kN + n) * kC) + c] = (bf16_t)v;
}

// ---------------------------------------------------------------------------
// K2: qT/kT (B,N,32) bf16:  qT[n,o] = sum_c x[c,n]*wq[o,c] + bq[o]
// One wave per (b, 16-row n-tile). A = xbT rows, B = wq/wk rows (f32->bf16).
// ---------------------------------------------------------------------------
__global__ void proj_qk_kernel(const bf16_t* __restrict__ xbT,
                               const float* __restrict__ wq, const float* __restrict__ bq,
                               const float* __restrict__ wk, const float* __restrict__ bk,
                               bf16_t* __restrict__ qT, bf16_t* __restrict__ kT) {
  int lane  = threadIdx.x & 31;
  int ntile = blockIdx.x;
  int b     = blockIdx.y;
  int nline = ntile * 16 + (lane & 15);
  const bf16_t* arow = xbT + (size_t)(b * kN + nline) * kC;
  int o = lane & 15;
  v8f qa0 = {}, qa1 = {}, ka0 = {}, ka1 = {};
  for (int k0 = 0; k0 < kC; k0 += 32) {
    v16bf a   = load_frag_k32(arow + k0, lane);
    v16bf bq0 = load_frag_k32_f32(wq + (size_t)o * kC + k0, lane);
    v16bf bq1 = load_frag_k32_f32(wq + (size_t)(o + 16) * kC + k0, lane);
    v16bf bk0 = load_frag_k32_f32(wk + (size_t)o * kC + k0, lane);
    v16bf bk1 = load_frag_k32_f32(wk + (size_t)(o + 16) * kC + k0, lane);
    qa0 = wmma_bf16(a, bq0, qa0);
    qa1 = wmma_bf16(a, bq1, qa1);
    ka0 = wmma_bf16(a, bk0, ka0);
    ka1 = wmma_bf16(a, bk1, ka1);
  }
  int hi = (lane >> 4) & 1;
  #pragma unroll
  for (int r = 0; r < 8; ++r) {
    int n = ntile * 16 + r + hi * 8;
    size_t base = (size_t)(b * kN + n) * kCQK;
    qT[base + o]      = (bf16_t)(qa0[r] + bq[o]);
    qT[base + o + 16] = (bf16_t)(qa1[r] + bq[o + 16]);
    kT[base + o]      = (bf16_t)(ka0[r] + bk[o]);
    kT[base + o + 16] = (bf16_t)(ka1[r] + bk[o + 16]);
  }
}

// ---------------------------------------------------------------------------
// K3: v (B,C,N) bf16: v[co,n] = sum_ci wv[co,ci]*x[ci,n] + bv[co]
// ---------------------------------------------------------------------------
__global__ void proj_v_kernel(const bf16_t* __restrict__ xbT,
                              const float* __restrict__ wv, const float* __restrict__ bv,
                              bf16_t* __restrict__ vmat) {
  int lane  = threadIdx.x & 31;
  int ntile = blockIdx.x, ctile = blockIdx.y, b = blockIdx.z;
  const float*  arow = wv  + (size_t)(ctile * 16 + (lane & 15)) * kC;
  const bf16_t* brow = xbT + (size_t)(b * kN + ntile * 16 + (lane & 15)) * kC;
  v8f acc = {};
  for (int k0 = 0; k0 < kC; k0 += 32) {
    v16bf a  = load_frag_k32_f32(arow + k0, lane);
    v16bf bb = load_frag_k32(brow + k0, lane);
    acc = wmma_bf16(a, bb, acc);
  }
  int hi = (lane >> 4) & 1;
  int n  = ntile * 16 + (lane & 15);
  #pragma unroll
  for (int r = 0; r < 8; ++r) {
    int cc = ctile * 16 + r + hi * 8;
    vmat[(size_t)(b * kC + cc) * kN + n] = (bf16_t)(acc[r] + bv[cc]);
  }
}

// ---------------------------------------------------------------------------
// K4: fused PAM attention.  One block (4 waves) per (b, 16-query tile).
// Pass 1: streaming row max / sumexp (online softmax stats).
// Pass 2: recompute energy per 64-key chunk, share normalized bf16 probs via
//         LDS, accumulate outT(16 x 256) with WMMA (each wave owns 64 chans).
// Epilogue: out = gamma_pam*pam + 2*x   (CAM kernel accumulates on top).
// ---------------------------------------------------------------------------
__global__ __launch_bounds__(128) void pam_kernel(
    const bf16_t* __restrict__ qT, const bf16_t* __restrict__ kT,
    const bf16_t* __restrict__ vmat, const float* __restrict__ x,
    const float* __restrict__ gamma_pam, float* __restrict__ out) {
  __shared__ bf16_t P[16][64];
  __shared__ float Msh[4][16], Lsh[4][16], Mfin[16], Linv[16];

  int lane = threadIdx.x & 31;
  int wave = threadIdx.x >> 5;
  int itile = blockIdx.x, b = blockIdx.y;
  int hi   = lane >> 4;
  int col  = lane & 15;

  // A fragment: q^T tile (16 queries x 32 ck), K contiguous in qT rows.
  v16bf aq = load_frag_k32(qT + (size_t)(b * kN + itile * 16 + col) * kCQK, lane);

  // ---- pass 1: per-row running (max, sumexp) over this wave's key range ----
  float m[8], l[8];
  #pragma unroll
  for (int r = 0; r < 8; ++r) { m[r] = -3.0e38f; l[r] = 0.f; }
  v8f zero = {};
  for (int t = 0; t < kN / 16 / 4; ++t) {
    int j0 = (t * 4 + wave) * 16;
    v16bf bk = load_frag_k32(kT + (size_t)(b * kN + j0 + col) * kCQK, lane);
    v8f e = wmma_bf16(aq, bk, zero);
    #pragma unroll
    for (int r = 0; r < 8; ++r) {
      float tm = rmax16(e[r]);
      float mn = fmaxf(m[r], tm);
      float s  = rsum16(__expf(e[r] - mn));
      l[r] = l[r] * __expf(m[r] - mn) + s;
      m[r] = mn;
    }
  }
  if (col == 0) {            // lanes 0 (rows 0-7) and 16 (rows 8-15)
    #pragma unroll
    for (int r = 0; r < 8; ++r) { Msh[wave][r + hi * 8] = m[r]; Lsh[wave][r + hi * 8] = l[r]; }
  }
  __syncthreads();
  if (threadIdx.x < 16) {
    int row = threadIdx.x;
    float M = -3.0e38f;
    for (int w = 0; w < 4; ++w) M = fmaxf(M, Msh[w][row]);
    float L = 0.f;
    for (int w = 0; w < 4; ++w) L += Lsh[w][row] * __expf(Msh[w][row] - M);
    Mfin[row] = M;
    Linv[row] = 1.f / L;
  }
  __syncthreads();
  float mr[8], li[8];
  #pragma unroll
  for (int r = 0; r < 8; ++r) { mr[r] = Mfin[r + hi * 8]; li[r] = Linv[r + hi * 8]; }

  // ---- pass 2: recompute probs per chunk, accumulate output GEMM ----
  v8f acc[4];
  #pragma unroll
  for (int cc = 0; cc < 4; ++cc) acc[cc] = zero;

  for (int jc = 0; jc < kN / 64; ++jc) {
    int j0c = jc * 64;
    int j0  = j0c + wave * 16;
    v16bf bk = load_frag_k32(kT + (size_t)(b * kN + j0 + col) * kCQK, lane);
    v8f e = wmma_bf16(aq, bk, zero);
    #pragma unroll
    for (int r = 0; r < 8; ++r) {
      float p = __expf(e[r] - mr[r]) * li[r];
      P[r + hi * 8][wave * 16 + col] = (bf16_t)p;   // C-layout -> LDS transpose
    }
    __syncthreads();

    // Rebuild P as two A fragments (K = 64 keys of this chunk).
    int ko = hi * 8;
    v16bf pa0, pa1;
    #pragma unroll
    for (int i2 = 0; i2 < 8; ++i2) {
      pa0[i2]     = P[col][ko + i2];
      pa0[i2 + 8] = P[col][ko + 16 + i2];
      pa1[i2]     = P[col][32 + ko + i2];
      pa1[i2 + 8] = P[col][32 + ko + 16 + i2];
    }
    #pragma unroll
    for (int cc = 0; cc < 4; ++cc) {
      int cline = (wave * 4 + cc) * 16 + col;
      const bf16_t* vrow = vmat + (size_t)(b * kC + cline) * kN + j0c;
      v16bf bv0 = load_frag_k32(vrow, lane);        // keys j0c .. j0c+31
      v16bf bv1 = load_frag_k32(vrow + 32, lane);   // keys j0c+32 .. j0c+63
      acc[cc] = wmma_bf16(pa0, bv0, acc[cc]);
      acc[cc] = wmma_bf16(pa1, bv1, acc[cc]);
    }
    __syncthreads();
  }

  // ---- epilogue: out = gamma_pam * pam + 2*x ----
  float gp = gamma_pam[0];
  #pragma unroll
  for (int cc = 0; cc < 4; ++cc) {
    int c = (wave * 4 + cc) * 16 + col;
    #pragma unroll
    for (int r = 0; r < 8; ++r) {
      int i = itile * 16 + r + hi * 8;
      size_t idx = (size_t)(b * kC + c) * kN + i;
      out[idx] = gp * acc[cc][r] + 2.f * x[idx];
    }
  }
}

// ---------------------------------------------------------------------------
// K5: CAM energy: ec[b,i,j] = sum_n x[i,n]*x[j,n]   (K = 4096)
// ---------------------------------------------------------------------------
__global__ void cam_energy_kernel(const bf16_t* __restrict__ xb, float* __restrict__ ec) {
  int lane = threadIdx.x & 31;
  int jt = blockIdx.x, it = blockIdx.y, b = blockIdx.z;
  const bf16_t* arow = xb + (size_t)(b * kC + it * 16 + (lane & 15)) * kN;
  const bf16_t* brow = xb + (size_t)(b * kC + jt * 16 + (lane & 15)) * kN;
  v8f acc = {};
  for (int k0 = 0; k0 < kN; k0 += 32)
    acc = wmma_bf16(load_frag_k32(arow + k0, lane), load_frag_k32(brow + k0, lane), acc);
  int hi = lane >> 4;
  #pragma unroll
  for (int r = 0; r < 8; ++r) {
    int i = it * 16 + r + hi * 8, j = jt * 16 + (lane & 15);
    ec[(size_t)(b * kC + i) * kC + j] = acc[r];
  }
}

// ---------------------------------------------------------------------------
// K6: CAM softmax of (rowmax - e):  p_j = exp(e_min - e_j) / sum.
// One wave per (b, row).
// ---------------------------------------------------------------------------
__global__ void cam_softmax_kernel(const float* __restrict__ ec, bf16_t* __restrict__ ac) {
  int i = blockIdx.x, b = blockIdx.y, lane = threadIdx.x;
  const float* row = ec + (size_t)(b * kC + i) * kC;
  float e[8], mn = 3.0e38f;
  #pragma unroll
  for (int t = 0; t < 8; ++t) { e[t] = row[lane + t * 32]; mn = fminf(mn, e[t]); }
  mn = rmin32(mn);
  float p[8], s = 0.f;
  #pragma unroll
  for (int t = 0; t < 8; ++t) { p[t] = __expf(mn - e[t]); s += p[t]; }
  s = rsum32(s);
  float inv = 1.f / s;
  bf16_t* orow = ac + (size_t)(b * kC + i) * kC;
  #pragma unroll
  for (int t = 0; t < 8; ++t) orow[lane + t * 32] = (bf16_t)(p[t] * inv);
}

// ---------------------------------------------------------------------------
// K7: CAM out: out[b,i,n] += gamma_cam * sum_j attn[i,j]*x[j,n]   (K = 256)
// ---------------------------------------------------------------------------
__global__ void cam_out_kernel(const bf16_t* __restrict__ ac, const bf16_t* __restrict__ xbT,
                               const float* __restrict__ gamma_cam, float* __restrict__ out) {
  int lane = threadIdx.x & 31;
  int nt = blockIdx.x, it = blockIdx.y, b = blockIdx.z;
  const bf16_t* arow = ac  + (size_t)(b * kC + it * 16 + (lane & 15)) * kC;
  const bf16_t* brow = xbT + (size_t)(b * kN + nt * 16 + (lane & 15)) * kC;
  v8f acc = {};
  for (int k0 = 0; k0 < kC; k0 += 32)
    acc = wmma_bf16(load_frag_k32(arow + k0, lane), load_frag_k32(brow + k0, lane), acc);
  float gc = gamma_cam[0];
  int hi = lane >> 4;
  #pragma unroll
  for (int r = 0; r < 8; ++r) {
    int i = it * 16 + r + hi * 8, n = nt * 16 + (lane & 15);
    size_t idx = (size_t)(b * kC + i) * kN + n;
    out[idx] += gc * acc[r];
  }
}

// ---------------------------------------------------------------------------
extern "C" void kernel_launch(void* const* d_in, const int* in_sizes, int n_in,
                              void* d_out, int out_size, void* d_ws, size_t ws_size,
                              hipStream_t stream) {
  (void)in_sizes; (void)n_in; (void)out_size; (void)ws_size;
  const float* x  = (const float*)d_in[0];
  const float* wq = (const float*)d_in[1];
  const float* bq = (const float*)d_in[2];
  const float* wk = (const float*)d_in[3];
  const float* bk = (const float*)d_in[4];
  const float* wv = (const float*)d_in[5];
  const float* bv = (const float*)d_in[6];
  const float* gp = (const float*)d_in[7];
  const float* gc = (const float*)d_in[8];
  float* out = (float*)d_out;

  char* ws = (char*)d_ws;
  size_t off = 0;
  auto carve = [&](size_t bytes) -> char* {
    char* p = ws + off;
    off += (bytes + 255) & ~(size_t)255;
    return p;
  };
  bf16_t* xb  = (bf16_t*)carve((size_t)kB * kC * kN * 2);   // 4 MB
  bf16_t* xbT = (bf16_t*)carve((size_t)kB * kN * kC * 2);   // 4 MB
  bf16_t* qT  = (bf16_t*)carve((size_t)kB * kN * kCQK * 2); // 512 KB
  bf16_t* kT  = (bf16_t*)carve((size_t)kB * kN * kCQK * 2); // 512 KB
  bf16_t* vm  = (bf16_t*)carve((size_t)kB * kC * kN * 2);   // 4 MB
  float*  ec  = (float*) carve((size_t)kB * kC * kC * 4);   // 512 KB
  bf16_t* ac  = (bf16_t*)carve((size_t)kB * kC * kC * 2);   // 256 KB

  convert_kernel<<<(kB * kC * kN) / 256, 256, 0, stream>>>(x, xb, xbT);
  proj_qk_kernel<<<dim3(kN / 16, kB), 32, 0, stream>>>(xbT, wq, bq, wk, bk, qT, kT);
  proj_v_kernel<<<dim3(kN / 16, kC / 16, kB), 32, 0, stream>>>(xbT, wv, bv, vm);
  pam_kernel<<<dim3(kN / 16, kB), 128, 0, stream>>>(qT, kT, vm, x, gp, out);
  cam_energy_kernel<<<dim3(kC / 16, kC / 16, kB), 32, 0, stream>>>(xb, ec);
  cam_softmax_kernel<<<dim3(kC, kB), 32, 0, stream>>>(ec, ac);
  cam_out_kernel<<<dim3(kN / 16, kC / 16, kB), 32, 0, stream>>>(ac, xbT, gc, out);
}
